// BridgeGCN_62345745268977
// MI455X (gfx1250) — matrix operand order repaired and checked by reading, not compiled, not run on hardware
//
#include <hip/hip_runtime.h>
#include <hip/hip_fp16.h>

typedef __attribute__((ext_vector_type(16))) _Float16 v16h;
typedef __attribute__((ext_vector_type(8)))  _Float16 h8;
typedef __attribute__((ext_vector_type(4)))  _Float16 h4;
typedef __attribute__((ext_vector_type(8)))  float    v8f;

// ---------------------------------------------------------------- utilities

__global__ void k_zero(float* p, size_t n) {
  size_t i = (size_t)blockIdx.x * blockDim.x + threadIdx.x;
  if (i < n) p[i] = 0.0f;
}

__global__ void k_init_deg(float* deg, int n) {
  int i = blockIdx.x * blockDim.x + threadIdx.x;
  if (i < n) deg[i] = 1.0f;                 // +1 self loop
}

__global__ void k_deg_accum(const int* __restrict__ ei, float* deg, int E) {
  int e = blockIdx.x * blockDim.x + threadIdx.x;
  if (e < E) atomicAdd(&deg[ei[E + e]], 1.0f);
}

__global__ void k_dinv(float* dinv, int n) {
  int i = blockIdx.x * blockDim.x + threadIdx.x;
  if (i < n) dinv[i] = rsqrtf(dinv[i]);
}

__global__ void k_tohalf(const float* __restrict__ a, _Float16* __restrict__ b, int n) {
  int i = blockIdx.x * blockDim.x + threadIdx.x;
  if (i < n) b[i] = (_Float16)a[i];
}

// ---------------------------------------------------------------- layer 1 linear (K=5)

__global__ void __launch_bounds__(256) k_lin1(const float* __restrict__ x,
                                              const float* __restrict__ W1,
                                              float* __restrict__ lin, int n) {
  __shared__ float sW[5 * 32];
  for (int i = threadIdx.x; i < 5 * 32; i += 256) sW[i] = W1[i];
  __syncthreads();
  int i = blockIdx.x * blockDim.x + threadIdx.x;
  if (i >= n) return;
  float xr[5];
#pragma unroll
  for (int k = 0; k < 5; k++) xr[k] = x[(size_t)i * 5 + k];
  float o[32];
#pragma unroll
  for (int c = 0; c < 32; c++) {
    float acc = 0.0f;
#pragma unroll
    for (int k = 0; k < 5; k++) acc += xr[k] * sW[k * 32 + c];
    o[c] = acc;
  }
  float* dst = lin + (size_t)i * 32;
#pragma unroll
  for (int q = 0; q < 8; q++) {
    float4 v; v.x = o[4*q]; v.y = o[4*q+1]; v.z = o[4*q+2]; v.w = o[4*q+3];
    *(float4*)(dst + q * 4) = v;
  }
}

// ---------------------------------------------------------------- GCN conv: self-loop init + bias

template <int F>
__global__ void k_conv_init(const float* __restrict__ lin, const float* __restrict__ bias,
                            const float* __restrict__ dinv, float* __restrict__ out, int n) {
  int i = blockIdx.x * blockDim.x + threadIdx.x;
  if (i >= n) return;
  float d = dinv[i];
  float d2 = d * d;
  const float4* L = (const float4*)(lin + (size_t)i * F);
  const float4* B = (const float4*)bias;
  float4* O = (float4*)(out + (size_t)i * F);
#pragma unroll
  for (int q = 0; q < F / 4; q++) {
    float4 v = L[q], b = B[q], r;
    r.x = v.x * d2 + b.x; r.y = v.y * d2 + b.y;
    r.z = v.z * d2 + b.z; r.w = v.w * d2 + b.w;
    O[q] = r;
  }
}

// ---------------------------------------------------------------- edge scatter-add

template <int F>
__global__ void k_edge_agg(const int* __restrict__ ei, const float* __restrict__ dinv,
                           const float* __restrict__ lin, float* __restrict__ out, int E) {
  constexpr int CH = F / 4;                 // float4 chunks per edge
  constexpr int SH = (CH == 8) ? 3 : 4;
  int tid = blockIdx.x * blockDim.x + threadIdx.x;
  if (tid >= E * CH) return;
  int e = tid >> SH;
  int q = tid & (CH - 1);
  int s = ei[e], d = ei[E + e];
  float c = dinv[s] * dinv[d];
  float4 v = *(const float4*)(lin + (size_t)s * F + q * 4);
  float* o = out + (size_t)d * F + q * 4;
  atomicAdd(o + 0, v.x * c);
  atomicAdd(o + 1, v.y * c);
  atomicAdd(o + 2, v.z * c);
  atomicAdd(o + 3, v.w * c);
}

// ---------------------------------------------------------------- batchnorm

template <int F>
__global__ void __launch_bounds__(256) k_bn_stats(const float* __restrict__ h,
                                                  float* __restrict__ stats, size_t total) {
  __shared__ float ss[256], sq[256];
  float s = 0.0f, s2 = 0.0f;
  // stride (grid*256) is a multiple of F, so each thread owns a fixed channel c = tid % F
  for (size_t e = (size_t)blockIdx.x * 256 + threadIdx.x; e < total; e += (size_t)gridDim.x * 256) {
    float v = h[e];
    s += v; s2 += v * v;
  }
  ss[threadIdx.x] = s; sq[threadIdx.x] = s2;
  __syncthreads();
#pragma unroll
  for (int off = 128; off >= F; off >>= 1) {
    if ((int)threadIdx.x < off) { ss[threadIdx.x] += ss[threadIdx.x + off]; sq[threadIdx.x] += sq[threadIdx.x + off]; }
    __syncthreads();
  }
  if ((int)threadIdx.x < F) {
    atomicAdd(&stats[threadIdx.x], ss[threadIdx.x]);
    atomicAdd(&stats[F + threadIdx.x], sq[threadIdx.x]);
  }
}

template <int F>
__global__ void k_bn_finish(float* stats, const float* __restrict__ g,
                            const float* __restrict__ be, float invN) {
  int c = threadIdx.x;
  if (c >= F) return;
  float mu  = stats[c] * invN;
  float var = stats[F + c] * invN - mu * mu;
  float a   = g[c] * rsqrtf(var + 1e-5f);
  stats[2 * F + c] = a;
  stats[3 * F + c] = be[c] - mu * a;
}

// vectorized: float4 in -> 4x f16 out, fused scale/shift + relu
template <int F>
__global__ void k_bn_apply(const float* __restrict__ h, const float* __restrict__ stats,
                           _Float16* __restrict__ o, size_t total4) {
  size_t t = (size_t)blockIdx.x * blockDim.x + threadIdx.x;
  if (t >= total4) return;
  size_t e = t * 4;
  int c = (int)(e & (size_t)(F - 1));
  float4 v = *(const float4*)(h + e);
  float a0 = stats[2 * F + c + 0], b0 = stats[3 * F + c + 0];
  float a1 = stats[2 * F + c + 1], b1 = stats[3 * F + c + 1];
  float a2 = stats[2 * F + c + 2], b2 = stats[3 * F + c + 2];
  float a3 = stats[2 * F + c + 3], b3 = stats[3 * F + c + 3];
  h4 r;
  r[0] = (_Float16)fmaxf(v.x * a0 + b0, 0.0f);
  r[1] = (_Float16)fmaxf(v.y * a1 + b1, 0.0f);
  r[2] = (_Float16)fmaxf(v.z * a2 + b2, 0.0f);
  r[3] = (_Float16)fmaxf(v.w * a3 + b3, 0.0f);
  *(h4*)(o + e) = r;
}

// ---------------------------------------------------------------- WMMA GEMM: [n x K](f16) @ [K x ND](f16) -> f32
// Weights are pre-swizzled into per-lane fragment layout in LDS, so each lane
// fetches a whole v16h B-fragment with contiguous 128-bit DS loads. B fragments
// are held in VGPRs and reused across TPW row tiles per wave.

template <int K, int ND, int TPW>
__global__ void __launch_bounds__(256) k_gemm_wmma(const _Float16* __restrict__ A,
                                                   const _Float16* __restrict__ W,
                                                   float* __restrict__ C, int nrows) {
  constexpr int NJ = ND / 16;               // column tiles
  constexpr int NK = K / 32;                // k steps
  constexpr int NFRAG = NJ * NK;
  __shared__ __align__(32) _Float16 lw[NFRAG * 32 * 16];

  // Cooperative fill in fragment layout:
  //   frag f = j*NK + kk ; lane<16: col=lane,   krow = kk*32 + t
  //                        lane>=16: col=lane-16, krow = kk*32 + 16 + t   (t = 0..15)
  for (int idx = threadIdx.x; idx < NFRAG * 32 * 16; idx += 256) {
    int t    = idx & 15;
    int ln   = (idx >> 4) & 31;
    int f    = idx >> 9;
    int kk   = f % NK;
    int j    = f / NK;
    int col  = ln & 15;
    int krow = kk * 32 + ((ln < 16) ? 0 : 16) + t;
    lw[idx] = W[krow * ND + j * 16 + col];
  }
  __syncthreads();

  int wave = threadIdx.x >> 5;
  int lane = threadIdx.x & 31;

  // B fragments -> registers (2x ds_load_b128 each)
  v16h bfr[NFRAG];
#pragma unroll
  for (int f = 0; f < NFRAG; f++)
    bfr[f] = *(const v16h*)(lw + (f * 32 + lane) * 16);

  // A fragment addressing: contiguous k = 16*(lane/16) + s, matching B fragments
  int rowA  = lane & 15;
  int koffA = (lane < 16) ? 0 : 16;
  // D layout: lanes 0-15: N=lane, vgpr r -> M=r; lanes 16-31: N=lane-16, vgpr r -> M=8+r
  int rbOff = (lane < 16) ? 0 : 8;
  int colD  = lane & 15;

  int tile0 = (blockIdx.x * 8 + wave) * TPW;
#pragma unroll
  for (int tt = 0; tt < TPW; tt++) {
    int r0 = (tile0 + tt) * 16;
    if (r0 >= nrows) break;                  // wave-uniform

    v16h afr[NK];
#pragma unroll
    for (int kk = 0; kk < NK; kk++) {
      const _Float16* p = A + (size_t)(r0 + rowA) * K + kk * 32 + koffA;
      h8 lo = *(const h8*)(p);
      h8 hi = *(const h8*)(p + 8);
      v16h a;
#pragma unroll
      for (int t = 0; t < 8; t++) { a[t] = lo[t]; a[8 + t] = hi[t]; }
      afr[kk] = a;
    }

#pragma unroll
    for (int j = 0; j < NJ; j++) {
      v8f acc = {};
#pragma unroll
      for (int kk = 0; kk < NK; kk++)
        acc = __builtin_amdgcn_wmma_f32_16x16x32_f16(false, afr[kk], false, bfr[j * NK + kk],
                                                     (short)0, acc, false, false);
      float* cp = C + (size_t)(r0 + rbOff) * ND + j * 16 + colD;
#pragma unroll
      for (int r = 0; r < 8; r++) cp[(size_t)r * ND] = acc[r];
    }
  }
}

// ---------------------------------------------------------------- pooling (relu folded in)

__global__ void k_pool(const float* __restrict__ h3, const int* __restrict__ batch,
                       float* __restrict__ pool, float* __restrict__ cnt, int n) {
  int tid = blockIdx.x * blockDim.x + threadIdx.x;
  int i = tid >> 4;
  int q = tid & 15;
  if (i >= n) return;
  int g = batch[i];
  if (q == 0) atomicAdd(&cnt[g], 1.0f);
  float4 v = *(const float4*)(h3 + (size_t)i * 64 + q * 4);
  float* p = pool + (size_t)g * 64 + q * 4;
  atomicAdd(p + 0, fmaxf(v.x, 0.0f));
  atomicAdd(p + 1, fmaxf(v.y, 0.0f));
  atomicAdd(p + 2, fmaxf(v.z, 0.0f));
  atomicAdd(p + 3, fmaxf(v.w, 0.0f));
}

// ---------------------------------------------------------------- head MLP: one graph per thread

__global__ void __launch_bounds__(256) k_head(const float* __restrict__ pool,
                                              const float* __restrict__ cnt,
                                              const float* __restrict__ fW1,
                                              const float* __restrict__ fb1,
                                              const float* __restrict__ fW2,
                                              const float* __restrict__ fb2,
                                              float* __restrict__ out, int G) {
  __shared__ float w1[64 * 32];
  __shared__ float b1s[32], w2[64], b2s[2];
  for (int i = threadIdx.x; i < 2048; i += 256) w1[i] = fW1[i];
  if (threadIdx.x < 32) b1s[threadIdx.x] = fb1[threadIdx.x];
  if (threadIdx.x < 64) w2[threadIdx.x] = fW2[threadIdx.x];
  if (threadIdx.x < 2)  b2s[threadIdx.x] = fb2[threadIdx.x];
  __syncthreads();
  int g = blockIdx.x * blockDim.x + threadIdx.x;
  if (g >= G) return;
  float inv = 1.0f / fmaxf(cnt[g], 1.0f);
  float p[64];
#pragma unroll
  for (int c = 0; c < 64; c++) p[c] = pool[(size_t)g * 64 + c] * inv;
  float o0 = b2s[0], o1 = b2s[1];
  for (int j = 0; j < 32; j++) {
    float z = b1s[j];
#pragma unroll
    for (int c = 0; c < 64; c++) z += p[c] * w1[c * 32 + j];
    z = fmaxf(z, 0.0f);
    o0 += z * w2[j * 2 + 0];
    o1 += z * w2[j * 2 + 1];
  }
  out[(size_t)g * 2 + 0] = o0;
  out[(size_t)g * 2 + 1] = o1;
}

// ---------------------------------------------------------------- launch

extern "C" void kernel_launch(void* const* d_in, const int* in_sizes, int n_in,
                              void* d_out, int out_size, void* d_ws, size_t ws_size,
                              hipStream_t stream) {
  const float* x    = (const float*)d_in[0];
  const int*   ei   = (const int*)  d_in[1];
  const int*   batc = (const int*)  d_in[2];
  const float* W1   = (const float*)d_in[3];
  const float* b1   = (const float*)d_in[4];
  const float* g1   = (const float*)d_in[5];
  const float* be1  = (const float*)d_in[6];
  const float* W2   = (const float*)d_in[7];
  const float* b2   = (const float*)d_in[8];
  const float* g2   = (const float*)d_in[9];
  const float* be2  = (const float*)d_in[10];
  const float* W3   = (const float*)d_in[11];
  const float* b3   = (const float*)d_in[12];
  const float* fW1  = (const float*)d_in[13];
  const float* fb1  = (const float*)d_in[14];
  const float* fW2  = (const float*)d_in[15];
  const float* fb2  = (const float*)d_in[16];
  float* out = (float*)d_out;
  (void)n_in; (void)ws_size;

  const int N = in_sizes[0] / 5;
  const int E = in_sizes[1] / 2;
  const int G = out_size / 2;

  char* ws = (char*)d_ws;
  size_t off = 0;
  auto alloc = [&](size_t bytes) -> char* {
    char* p = ws + off;
    off = (off + bytes + 255) & ~(size_t)255;
    return p;
  };

  float*    dinv  = (float*)   alloc((size_t)N * 4);
  float*    stats = (float*)   alloc(1024);
  _Float16* w2h   = (_Float16*)alloc(32 * 64 * 2);
  _Float16* w3h   = (_Float16*)alloc(64 * 64 * 2);
  float*    lin32 = (float*)   alloc((size_t)N * 32 * 4);
  float*    out32 = (float*)   alloc((size_t)N * 32 * 4);
  _Float16* h16a  = (_Float16*)alloc((size_t)N * 32 * 2);
  float*    lin64 = (float*)   alloc((size_t)N * 64 * 4);   // reused for layer-3 lin
  float*    out64 = (float*)   alloc((size_t)N * 64 * 4);   // reused for layer-3 out
  _Float16* h16b  = (_Float16*)alloc((size_t)N * 64 * 2);
  float*    pool  = (float*)   alloc((size_t)G * 64 * 4);
  float*    cnt   = (float*)   alloc((size_t)G * 4);

  const int nbN = (N + 255) / 256;
  const int nbE = (E + 255) / 256;
  constexpr int TPW = 4;                       // row tiles per wave
  const int tiles = (N + 15) / 16;
  const int gemmBlocks = (tiles + 8 * TPW - 1) / (8 * TPW);

  // degree / dinv
  k_init_deg<<<nbN, 256, 0, stream>>>(dinv, N);
  k_deg_accum<<<nbE, 256, 0, stream>>>(ei, dinv, E);
  k_dinv<<<nbN, 256, 0, stream>>>(dinv, N);

  // weight conversion for WMMA layers
  k_tohalf<<<8, 256, 0, stream>>>(W2, w2h, 32 * 64);
  k_tohalf<<<16, 256, 0, stream>>>(W3, w3h, 64 * 64);

  // ---- layer 1: conv(5->32)
  k_lin1<<<nbN, 256, 0, stream>>>(x, W1, lin32, N);
  k_conv_init<32><<<nbN, 256, 0, stream>>>(lin32, b1, dinv, out32, N);
  k_edge_agg<32><<<(E * 8 + 255) / 256, 256, 0, stream>>>(ei, dinv, lin32, out32, E);

  // BN1 + relu -> f16
  k_zero<<<1, 256, 0, stream>>>(stats, 256);
  k_bn_stats<32><<<2048, 256, 0, stream>>>(out32, stats, (size_t)N * 32);
  k_bn_finish<32><<<1, 32, 0, stream>>>(stats, g1, be1, 1.0f / (float)N);
  k_bn_apply<32><<<(int)(((size_t)N * 8 + 255) / 256), 256, 0, stream>>>(out32, stats, h16a, (size_t)N * 8);

  // ---- layer 2: conv(32->64) with WMMA linear
  k_gemm_wmma<32, 64, TPW><<<gemmBlocks, 256, 0, stream>>>(h16a, w2h, lin64, N);
  k_conv_init<64><<<nbN, 256, 0, stream>>>(lin64, b2, dinv, out64, N);
  k_edge_agg<64><<<(E * 16 + 255) / 256, 256, 0, stream>>>(ei, dinv, lin64, out64, E);

  // BN2 + relu -> f16
  k_zero<<<1, 256, 0, stream>>>(stats, 256);
  k_bn_stats<64><<<2048, 256, 0, stream>>>(out64, stats, (size_t)N * 64);
  k_bn_finish<64><<<1, 64, 0, stream>>>(stats, g2, be2, 1.0f / (float)N);
  k_bn_apply<64><<<(int)(((size_t)N * 16 + 255) / 256), 256, 0, stream>>>(out64, stats, h16b, (size_t)N * 16);

  // ---- layer 3: conv(64->64) with WMMA linear (reuse lin64/out64)
  k_gemm_wmma<64, 64, TPW><<<gemmBlocks, 256, 0, stream>>>(h16b, w3h, lin64, N);
  k_conv_init<64><<<nbN, 256, 0, stream>>>(lin64, b3, dinv, out64, N);
  k_edge_agg<64><<<(E * 16 + 255) / 256, 256, 0, stream>>>(ei, dinv, lin64, out64, E);

  // ---- mean pool (relu at read) + head MLP
  k_zero<<<(int)(((size_t)G * 64 + 255) / 256), 256, 0, stream>>>(pool, (size_t)G * 64);
  k_zero<<<(G + 255) / 256, 256, 0, stream>>>(cnt, (size_t)G);
  k_pool<<<(N * 16 + 255) / 256, 256, 0, stream>>>(out64, batc, pool, cnt, N);
  k_head<<<(G + 255) / 256, 256, 0, stream>>>(pool, cnt, fW1, fb1, fW2, fb2, out, G);
}
